// MultiHeadAttention_27419071218232
// MI455X (gfx1250) — matrix operand order repaired and checked by reading, not compiled, not run on hardware
//
#include <hip/hip_runtime.h>
#include <hip/hip_bf16.h>

// ---------------- problem constants ----------------
constexpr int B_ = 2;
constexpr int T_ = 2048;
constexpr int E_ = 1024;
constexpr int H_ = 16;
constexpr int D_ = 64;      // head dim
constexpr int M_ = B_ * T_; // 4096 rows for the projection GEMMs

typedef __bf16 bf16;
typedef __attribute__((ext_vector_type(8)))  __bf16 v8bf;
typedef __attribute__((ext_vector_type(16))) __bf16 v16bf;
typedef __attribute__((ext_vector_type(8)))  float  v8f;

__device__ __forceinline__ v16bf cat8(v8bf lo, v8bf hi) {
    return __builtin_shufflevector(lo, hi, 0,1,2,3,4,5,6,7,8,9,10,11,12,13,14,15);
}

__device__ __forceinline__ v8f wmma_bf16(v16bf a, v16bf b, v8f c) {
    // D = A(16x32 bf16) * B(32x16 bf16) + C(16x16 f32)
    return __builtin_amdgcn_wmma_f32_16x16x32_bf16(
        /*neg_a=*/false, a, /*neg_b=*/false, b,
        /*c_mod=*/(short)0, c, /*reuse_a=*/false, /*reuse_b=*/false);
}

// A-frag 16x32 bf16: lane half 0 holds K 0-15, half 1 holds K 16-31
__device__ __forceinline__ v16bf load_afrag(const bf16* row_base, int k0, int hl) {
    v8bf lo = *(const v8bf*)(row_base + k0 + hl * 16);
    v8bf hi = *(const v8bf*)(row_base + k0 + hl * 16 + 8);
    return cat8(lo, hi);
}

// ---------------- f32 -> bf16 convert ----------------
__global__ void cvt_f32_bf16(const float* __restrict__ src, bf16* __restrict__ dst, int n) {
    int i = blockIdx.x * blockDim.x + threadIdx.x;
    if (i < n) dst[i] = (bf16)src[i];
}

// ============================================================================
// Fused QKV projection: C = x @ W^T + bias
//   x:[M_,E_] bf16 row-major, W:[E_,E_] bf16 row-major.
//   blockIdx.z: 0 -> Q [B,H,T,D], 1 -> K [B,H,T,D], 2 -> V transposed [B,H,D,T]
// Each wave computes a 32x64 tile (2 A-frags x 4 B-frags = 8 WMMA per k-step),
// software-pipelined: next k-step fragments load while current WMMAs execute.
// ============================================================================
__global__ __launch_bounds__(128) void gemm_qkv(
    const bf16* __restrict__ xb,
    const bf16* __restrict__ Wqb, const float* __restrict__ bq,
    const bf16* __restrict__ Wkb, const float* __restrict__ bk,
    const bf16* __restrict__ Wvb, const float* __restrict__ bv,
    bf16* __restrict__ Qout, bf16* __restrict__ Kout, bf16* __restrict__ Vtout)
{
    const int wave = threadIdx.x >> 5;
    const int lane = threadIdx.x & 31;
    const int hl   = lane >> 4;
    const int r    = lane & 15;
    const int m0 = blockIdx.x * 128 + wave * 32;   // 32 rows per wave
    const int n0 = blockIdx.y * 64;
    const int z  = blockIdx.z;

    const bf16*  W    = (z == 0) ? Wqb : (z == 1) ? Wkb : Wvb;
    const float* bias = (z == 0) ? bq  : (z == 1) ? bk  : bv;

    const bf16* arow0 = xb + (size_t)(m0 + r) * E_;        // rows m0..m0+15
    const bf16* arow1 = arow0 + (size_t)16 * E_;           // rows m0+16..m0+31
    const bf16* wrow[4];
#pragma unroll
    for (int f = 0; f < 4; ++f)
        wrow[f] = W + (size_t)(n0 + f * 16 + r) * E_ + hl * 16;

    v8f acc[8] = {};   // [0..3]: rows m0..+15 x ntile f ; [4..7]: rows m0+16..+31

    // prologue: fragments for k = 0
    v16bf a0 = load_afrag(arow0, 0, hl);
    v16bf a1 = load_afrag(arow1, 0, hl);
    v16bf b0 = *(const v16bf*)(wrow[0]);
    v16bf b1 = *(const v16bf*)(wrow[1]);
    v16bf b2 = *(const v16bf*)(wrow[2]);
    v16bf b3 = *(const v16bf*)(wrow[3]);

    for (int k0 = 0; k0 < E_; k0 += 32) {
        const int kn = (k0 + 32) & (E_ - 1);   // wraps to 0 on last iter (discarded)
        // issue next-step loads first so they overlap the WMMAs below
        v16bf na0 = load_afrag(arow0, kn, hl);
        v16bf na1 = load_afrag(arow1, kn, hl);
        v16bf nb0 = *(const v16bf*)(wrow[0] + kn);
        v16bf nb1 = *(const v16bf*)(wrow[1] + kn);
        v16bf nb2 = *(const v16bf*)(wrow[2] + kn);
        v16bf nb3 = *(const v16bf*)(wrow[3] + kn);

        acc[0] = wmma_bf16(a0, b0, acc[0]);
        acc[1] = wmma_bf16(a0, b1, acc[1]);
        acc[2] = wmma_bf16(a0, b2, acc[2]);
        acc[3] = wmma_bf16(a0, b3, acc[3]);
        acc[4] = wmma_bf16(a1, b0, acc[4]);
        acc[5] = wmma_bf16(a1, b1, acc[5]);
        acc[6] = wmma_bf16(a1, b2, acc[6]);
        acc[7] = wmma_bf16(a1, b3, acc[7]);

        a0 = na0; a1 = na1; b0 = nb0; b1 = nb1; b2 = nb2; b3 = nb3;
    }

#pragma unroll
    for (int g = 0; g < 2; ++g) {          // A row-block
#pragma unroll
        for (int f = 0; f < 4; ++f) {      // N tile
            const int n = n0 + f * 16 + r;
            const float bv_ = bias[n];
            const int h = n >> 6;
            const int d = n & 63;
#pragma unroll
            for (int rr = 0; rr < 8; ++rr) {
                const int m = m0 + g * 16 + rr + 8 * hl;
                const int b = m >> 11;             // m / T_
                const int t = m & (T_ - 1);
                const float val = acc[g * 4 + f][rr] + bv_;
                if (z == 2) {
                    Vtout[(((size_t)(b * H_ + h)) * D_ + d) * T_ + t] = (bf16)val;
                } else {
                    bf16* dst = (z == 0) ? Qout : Kout;
                    dst[(((size_t)(b * H_ + h)) * T_ + t) * D_ + d] = (bf16)val;
                }
            }
        }
    }
}

// ============================================================================
// Flash attention (online softmax).
// Q,K: [B,H,T,D] bf16 ; Vt: [B,H,D,T] bf16 ; out: [B,T,E] bf16 (= [B,T,H,D])
// V-fragment loads are issued before the softmax so they overlap the exp chain.
// ============================================================================
__global__ __launch_bounds__(256) void attn_flash(
    const bf16* __restrict__ Qb, const bf16* __restrict__ Kb,
    const bf16* __restrict__ Vtb, bf16* __restrict__ attnout)
{
    constexpr int WAVES = 8;
    __shared__ float sS[WAVES][16][32];              // raw scores (f32, C-layout dump)
    __shared__ bf16  sP[WAVES][16][32];              // exp'd probs (bf16, A-layout source)
    __shared__ alignas(32) float sScale[WAVES][16];  // per-row rescale broadcast

    const int wave = threadIdx.x >> 5;
    const int lane = threadIdx.x & 31;
    const int hl   = lane >> 4;
    const int r    = lane & 15;
    const int b = blockIdx.z;
    const int h = blockIdx.y;
    const int q0 = blockIdx.x * (16 * WAVES) + wave * 16;

    const bf16* Qp = Qb  + (((size_t)(b * H_ + h)) * T_ + q0) * D_;
    const bf16* Kp = Kb  + ((size_t)(b * H_ + h)) * T_ * D_;
    const bf16* Vp = Vtb + ((size_t)(b * H_ + h)) * D_ * T_;

    float (*S)[32] = sS[wave];
    bf16  (*P)[32] = sP[wave];
    float *Scale   = sScale[wave];

    // Preload Q fragments for this wave's 16 rows: d 0-31 and d 32-63
    const bf16* qrow = Qp + (size_t)r * D_;
    v16bf qf0 = cat8(*(const v8bf*)(qrow + hl * 16),      *(const v8bf*)(qrow + 8 + hl * 16));
    v16bf qf1 = cat8(*(const v8bf*)(qrow + 32 + hl * 16), *(const v8bf*)(qrow + 40 + hl * 16));

    const bf16* vrow[4];
#pragma unroll
    for (int f = 0; f < 4; ++f)
        vrow[f] = Vp + (size_t)(f * 16 + r) * T_ + hl * 16;

    v8f o[4] = {};
    float run_max = -3.0e38f;
    float run_sum = 0.0f;

    for (int kk0 = 0; kk0 < T_; kk0 += 32) {
        // ---- scores S = (Q K^T) * 1/sqrt(D) for 32 keys ----
#pragma unroll
        for (int j = 0; j < 2; ++j) {
            const bf16* krow = Kp + (size_t)(kk0 + j * 16 + r) * D_;
            v16bf kb0 = *(const v16bf*)(krow + hl * 16);        // d 0-31 chunk
            v16bf kb1 = *(const v16bf*)(krow + 32 + hl * 16);   // d 32-63 chunk
            __builtin_prefetch(krow + 32 * D_, 0, 1);
            v8f s = {};
            s = wmma_bf16(qf0, kb0, s);
            s = wmma_bf16(qf1, kb1, s);
#pragma unroll
            for (int rr = 0; rr < 8; ++rr)
                S[rr + 8 * hl][j * 16 + r] = s[rr] * 0.125f;    // 1/sqrt(64)
        }

        // ---- issue V-fragment loads now; they overlap the softmax below ----
        v16bf vb[4];
#pragma unroll
        for (int f = 0; f < 4; ++f)
            vb[f] = *(const v16bf*)(vrow[f] + kk0);

        asm volatile("s_wait_dscnt 0" ::: "memory");

        // ---- online softmax: lanes 0-15 own one row each ----
        if (hl == 0) {
            float tmax = -3.0e38f;
#pragma unroll
            for (int c = 0; c < 32; ++c) tmax = fmaxf(tmax, S[r][c]);
            const float mnew = fmaxf(run_max, tmax);
            const float corr = __expf(run_max - mnew);
            float ssum = 0.0f;
#pragma unroll
            for (int c = 0; c < 32; ++c) {
                const float p = __expf(S[r][c] - mnew);
                ssum += p;
                P[r][c] = (bf16)p;
            }
            run_sum = run_sum * corr + ssum;
            run_max = mnew;
            Scale[r] = corr;
        }
        asm volatile("s_wait_dscnt 0" ::: "memory");

        // ---- rescale accumulators: row m = rr + 8*hl ----
        v8f corr8 = *(const v8f*)&Scale[8 * hl];
#pragma unroll
        for (int f = 0; f < 4; ++f)
#pragma unroll
            for (int rr = 0; rr < 8; ++rr)
                o[f][rr] *= corr8[rr];

        // ---- O += P @ V  (P 16x32 A-frag from LDS, V B-frags preloaded) ----
        v16bf pf = cat8(*(const v8bf*)&P[r][hl * 16], *(const v8bf*)&P[r][8 + hl * 16]);
#pragma unroll
        for (int f = 0; f < 4; ++f)
            o[f] = wmma_bf16(pf, vb[f], o[f]);
    }

    // ---- finalize: divide by row sum, store as [B,T,H*D] bf16 ----
    if (hl == 0) Scale[r] = 1.0f / run_sum;
    asm volatile("s_wait_dscnt 0" ::: "memory");
    v8f inv8 = *(const v8f*)&Scale[8 * hl];
#pragma unroll
    for (int f = 0; f < 4; ++f) {
#pragma unroll
        for (int rr = 0; rr < 8; ++rr) {
            const int m = q0 + rr + 8 * hl;   // query row within [T]
            const float val = o[f][rr] * inv8[rr];
            attnout[((size_t)(b * T_ + m)) * E_ + h * 64 + f * 16 + r] = (bf16)val;
        }
    }
}

// ============================================================================
// Output projection: out = attn @ Wo^T + bo (f32 out), 32x64 tile, pipelined.
// ============================================================================
__global__ __launch_bounds__(128) void gemm_out(
    const bf16* __restrict__ A, const bf16* __restrict__ Wob,
    const float* __restrict__ bo, float* __restrict__ out)
{
    const int wave = threadIdx.x >> 5;
    const int lane = threadIdx.x & 31;
    const int hl   = lane >> 4;
    const int r    = lane & 15;
    const int m0 = blockIdx.x * 128 + wave * 32;
    const int n0 = blockIdx.y * 64;

    const bf16* arow0 = A + (size_t)(m0 + r) * E_;
    const bf16* arow1 = arow0 + (size_t)16 * E_;
    const bf16* wrow[4];
#pragma unroll
    for (int f = 0; f < 4; ++f)
        wrow[f] = Wob + (size_t)(n0 + f * 16 + r) * E_ + hl * 16;

    v8f acc[8] = {};

    v16bf a0 = load_afrag(arow0, 0, hl);
    v16bf a1 = load_afrag(arow1, 0, hl);
    v16bf b0 = *(const v16bf*)(wrow[0]);
    v16bf b1 = *(const v16bf*)(wrow[1]);
    v16bf b2 = *(const v16bf*)(wrow[2]);
    v16bf b3 = *(const v16bf*)(wrow[3]);

    for (int k0 = 0; k0 < E_; k0 += 32) {
        const int kn = (k0 + 32) & (E_ - 1);
        v16bf na0 = load_afrag(arow0, kn, hl);
        v16bf na1 = load_afrag(arow1, kn, hl);
        v16bf nb0 = *(const v16bf*)(wrow[0] + kn);
        v16bf nb1 = *(const v16bf*)(wrow[1] + kn);
        v16bf nb2 = *(const v16bf*)(wrow[2] + kn);
        v16bf nb3 = *(const v16bf*)(wrow[3] + kn);

        acc[0] = wmma_bf16(a0, b0, acc[0]);
        acc[1] = wmma_bf16(a0, b1, acc[1]);
        acc[2] = wmma_bf16(a0, b2, acc[2]);
        acc[3] = wmma_bf16(a0, b3, acc[3]);
        acc[4] = wmma_bf16(a1, b0, acc[4]);
        acc[5] = wmma_bf16(a1, b1, acc[5]);
        acc[6] = wmma_bf16(a1, b2, acc[6]);
        acc[7] = wmma_bf16(a1, b3, acc[7]);

        a0 = na0; a1 = na1; b0 = nb0; b1 = nb1; b2 = nb2; b3 = nb3;
    }

#pragma unroll
    for (int g = 0; g < 2; ++g) {
#pragma unroll
        for (int f = 0; f < 4; ++f) {
            const int n = n0 + f * 16 + r;
            const float bv_ = bo[n];
#pragma unroll
            for (int rr = 0; rr < 8; ++rr) {
                const int m = m0 + g * 16 + rr + 8 * hl;
                out[(size_t)m * E_ + n] = acc[g * 4 + f][rr] + bv_;
            }
        }
    }
}

// ---------------- host launcher ----------------
extern "C" void kernel_launch(void* const* d_in, const int* in_sizes, int n_in,
                              void* d_out, int out_size, void* d_ws, size_t ws_size,
                              hipStream_t stream) {
    (void)in_sizes; (void)n_in; (void)out_size; (void)ws_size;
    const float* x  = (const float*)d_in[0];
    const float* Wq = (const float*)d_in[1];
    const float* bq = (const float*)d_in[2];
    const float* Wk = (const float*)d_in[3];
    const float* bk = (const float*)d_in[4];
    const float* Wv = (const float*)d_in[5];
    const float* bv = (const float*)d_in[6];
    const float* Wo = (const float*)d_in[7];
    const float* bo = (const float*)d_in[8];
    float* out = (float*)d_out;

    char* ws = (char*)d_ws;
    size_t off = 0;
    auto alloc = [&](size_t bytes) -> void* {
        void* p = ws + off;
        off += (bytes + 255) & ~(size_t)255;
        return p;
    };
    const size_t nx = (size_t)M_ * E_;   // 4M elems
    const size_t nw = (size_t)E_ * E_;   // 1M elems
    bf16* xb    = (bf16*)alloc(nx * 2);
    bf16* Wqb   = (bf16*)alloc(nw * 2);
    bf16* Wkb   = (bf16*)alloc(nw * 2);
    bf16* Wvb   = (bf16*)alloc(nw * 2);
    bf16* Wob   = (bf16*)alloc(nw * 2);
    bf16* Qb    = (bf16*)alloc(nx * 2);  // [B,H,T,D]
    bf16* Kb    = (bf16*)alloc(nx * 2);  // [B,H,T,D]
    bf16* Vtb   = (bf16*)alloc(nx * 2);  // [B,H,D,T]
    bf16* attnb = (bf16*)alloc(nx * 2);  // [B,T,E]

    cvt_f32_bf16<<<(unsigned)((nx + 255) / 256), 256, 0, stream>>>(x,  xb,  (int)nx);
    cvt_f32_bf16<<<(unsigned)((nw + 255) / 256), 256, 0, stream>>>(Wq, Wqb, (int)nw);
    cvt_f32_bf16<<<(unsigned)((nw + 255) / 256), 256, 0, stream>>>(Wk, Wkb, (int)nw);
    cvt_f32_bf16<<<(unsigned)((nw + 255) / 256), 256, 0, stream>>>(Wv, Wvb, (int)nw);
    cvt_f32_bf16<<<(unsigned)((nw + 255) / 256), 256, 0, stream>>>(Wo, Wob, (int)nw);

    gemm_qkv<<<dim3(M_ / 128, E_ / 64, 3), 128, 0, stream>>>(
        xb, Wqb, bq, Wkb, bk, Wvb, bv, Qb, Kb, Vtb);

    attn_flash<<<dim3(T_ / 128, H_, B_), 256, 0, stream>>>(Qb, Kb, Vtb, attnb);

    gemm_out<<<dim3(M_ / 128, E_ / 64), 128, 0, stream>>>(attnb, Wob, bo, out);
}
